// RelationEncoder_57561151701198
// MI455X (gfx1250) — matrix operand order repaired and checked by reading, not compiled, not run on hardware
//
#include <hip/hip_runtime.h>

typedef __attribute__((ext_vector_type(16))) __bf16 v16bf;
typedef __attribute__((ext_vector_type(8)))  float  v8f;

union FragU { uint4 u4[2]; v16bf v; };

// Native hardware convert (RNE) instead of integer bit-twiddling.
__device__ __forceinline__ __bf16 f2bf(float x) { return (__bf16)x; }

// Pack two floats into one dword of bf16x2 (backend can fuse to v_cvt_pk_bf16_f32).
__device__ __forceinline__ unsigned int pack2bf(float a, float b) {
  union { __bf16 h[2]; unsigned int u; } x;
  x.h[0] = (__bf16)a;
  x.h[1] = (__bf16)b;
  return x.u;
}

// ---------------- weight fp32 -> bf16 conversion pre-pass ----------------
__global__ __launch_bounds__(256) void cvt_bf16(const float* __restrict__ src,
                                                __bf16* __restrict__ dst, int n) {
  int i = blockIdx.x * blockDim.x + threadIdx.x;
  int stride = gridDim.x * blockDim.x;
  for (; i < n; i += stride) dst[i] = (__bf16)src[i];
}

// ---------------- WMMA fragment helpers ----------------
// A fragment (16x32, MxK, bf16): lane L holds row M = L%16;
//   K = (L/16)*8 + {0..7} in v[0..3], K = (L/16)*8 + 16 + {0..7} in v[4..7].
__device__ __forceinline__ v16bf loadA(const __bf16* X, int mbase, int k0, int lane) {
  const __bf16* p = X + (mbase + (lane & 15)) * 256 + k0 + ((lane >> 4) << 3);
  FragU f;
  f.u4[0] = *(const uint4*)p;          // K + 0..7
  f.u4[1] = *(const uint4*)(p + 16);   // K + 16..23
  return f.v;
}
// B fragment (32x16, KxN, bf16): lane L holds column N = n0+L%16;
//   K = (L/16)*8 + {0..7} in v[0..3], K = (L/16)*8 + 16 + {0..7} in v[4..7] (row-major W, ld = 256).
__device__ __forceinline__ v16bf loadB(const __bf16* __restrict__ W, int k0, int n0, int lane) {
  const __bf16* p = W + (k0 + ((lane >> 4) << 3)) * 256 + n0 + (lane & 15);
  v16bf v;
#pragma unroll
  for (int s = 0; s < 8; ++s) {
    v[s] = p[s * 256];
    v[s + 8] = p[(s + 16) * 256];
  }
  return v;
}
__device__ __forceinline__ v8f wmma_bf16(v16bf a, v16bf b, v8f c) {
  return __builtin_amdgcn_wmma_f32_16x16x32_bf16(false, a, false, b, (short)0, c, false, false);
}

// Epilogue: D tile layout (f32 16x16): VGPR r, lanes 0-15 -> M=r, N=lane; lanes 16-31 -> M=8+r.
__device__ __forceinline__ void store_h(__bf16* Hb, v8f acc, int mbase, int n0,
                                        int lane, float bias, bool relu) {
  int n = n0 + (lane & 15);
  int moff = mbase + ((lane >> 4) << 3);
#pragma unroll
  for (int r = 0; r < 8; ++r) {
    float v = acc[r] + bias;
    if (relu) v = fmaxf(v, 0.0f);
    Hb[(moff + r) * 256 + n] = (__bf16)v;
  }
}
__device__ __forceinline__ void store_h_dist(__bf16* Hb, v8f acc, int mbase, int n0,
                                             int lane, float bias, const float* dist_s, float wlast) {
  int n = n0 + (lane & 15);
  int moff = mbase + ((lane >> 4) << 3);
#pragma unroll
  for (int r = 0; r < 8; ++r) {
    float v = fmaf(dist_s[moff + r], wlast, acc[r] + bias);  // rank-1 distance column
    v = fmaxf(v, 0.0f);
    Hb[(moff + r) * 256 + n] = (__bf16)v;
  }
}

// ---------------- fused relation-encoder kernel ----------------
// Block: 512 threads = 16 waves. 32 pairs (same b,i; j = j0..j0+31), full H=256 output.
// Wave w owns N-tile [w*16, w*16+16) and both M-tiles (rows 0-15, 16-31).
__global__ __launch_bounds__(512, 1) void relenc_wmma(
    const float* __restrict__ features, const float* __restrict__ distances,
    const float* __restrict__ mask, const float* __restrict__ Ws1_f32,
    const float* __restrict__ bs1, const float* __restrict__ bs2,
    const float* __restrict__ bt1, const float* __restrict__ bt2,
    const float* __restrict__ bi1, const float* __restrict__ bi2,
    const float* __restrict__ bfin,
    const __bf16* __restrict__ Wbf,
    float* __restrict__ out) {
  const __bf16* W1s = Wbf + 0;        // 256x256 bf16 (rows 0..255 of Ws1)
  const __bf16* W1t = Wbf + 65536;
  const __bf16* W1i = Wbf + 131072;
  const __bf16* W2s = Wbf + 196608;
  const __bf16* W2t = Wbf + 262144;
  const __bf16* W2i = Wbf + 327680;
  const __bf16* WfA = Wbf + 393216;   // Wf rows   0..255
  const __bf16* WfB = Wbf + 458752;   // Wf rows 256..511
  const __bf16* WfC = Wbf + 524288;   // Wf rows 512..767

  extern __shared__ __align__(16) char smem[];
  float* fi_s   = (float*)smem;                 // 128
  float* dist_s = fi_s + 128;                   // 32
  float* mp_s   = dist_s + 32;                  // 32
  __bf16* X1 = (__bf16*)(mp_s + 32);            // [32][256] bf16: [fi | fj]
  __bf16* X2 = X1 + 8192;                       // [fi*fj | fi+fj]
  __bf16* Hs = X2 + 8192;                       // layer-1 hiddens
  __bf16* Ht = Hs + 8192;
  __bf16* Hi = Ht + 8192;
  __bf16* Gi = Hi + 8192;                       // layer-2 interaction out
  __bf16* Gs = X1;                              // reuse (X dead after L1)
  __bf16* Gt = X2;

  int bid = blockIdx.x;
  int jt = bid % 3;
  int i  = (bid / 3) % 96;
  int b  = bid / (3 * 96);
  int j0 = jt * 32;

  int tid  = threadIdx.x;
  int lane = tid & 31;
  int wave = tid >> 5;
  int n0   = wave * 16;

  if (tid < 128)      fi_s[tid] = features[(b * 96 + i) * 128 + tid];
  else if (tid < 160) dist_s[tid - 128] = distances[(b * 96 + i) * 96 + j0 + (tid - 128)];
  else if (tid < 192) mp_s[tid - 160] = mask[b * 96 + i] * mask[b * 96 + j0 + (tid - 160)];
  __syncthreads();

  // Stage activations: 2 adjacent k per thread -> packed bf16x2 dword stores.
  for (int idx = tid; idx < 32 * 64; idx += 512) {
    int m = idx >> 6, kp = (idx & 63) << 1;      // kp = even k
    const float2 fj2 = *(const float2*)&features[(b * 96 + j0 + m) * 128 + kp];
    const float2 fi2 = *(const float2*)&fi_s[kp];
    unsigned int* x1a = (unsigned int*)&X1[m * 256 + kp];
    unsigned int* x1b = (unsigned int*)&X1[m * 256 + 128 + kp];
    unsigned int* x2a = (unsigned int*)&X2[m * 256 + kp];
    unsigned int* x2b = (unsigned int*)&X2[m * 256 + 128 + kp];
    *x1a = pack2bf(fi2.x, fi2.y);
    *x1b = pack2bf(fj2.x, fj2.y);
    *x2a = pack2bf(fi2.x * fj2.x, fi2.y * fj2.y);
    *x2b = pack2bf(fi2.x + fj2.x, fi2.y + fj2.y);
  }
  __syncthreads();

  // ---- layer 1: three GEMMs, shared B loads feed 2 M-tiles each ----
  {
    v8f s0 = {}, s1 = {}, t0 = {}, t1 = {}, p0 = {}, p1 = {};
#pragma unroll
    for (int kc = 0; kc < 8; ++kc) {
      int k0 = kc * 32;
      v16bf Bs = loadB(W1s, k0, n0, lane);
      v16bf Bt = loadB(W1t, k0, n0, lane);
      v16bf Bi = loadB(W1i, k0, n0, lane);
      v16bf A0 = loadA(X1, 0,  k0, lane);
      v16bf A1 = loadA(X1, 16, k0, lane);
      v16bf C0 = loadA(X2, 0,  k0, lane);
      v16bf C1 = loadA(X2, 16, k0, lane);
      s0 = wmma_bf16(A0, Bs, s0); s1 = wmma_bf16(A1, Bs, s1);
      t0 = wmma_bf16(A0, Bt, t0); t1 = wmma_bf16(A1, Bt, t1);
      p0 = wmma_bf16(C0, Bi, p0); p1 = wmma_bf16(C1, Bi, p1);
    }
    int ncol = n0 + (lane & 15);
    float wl = Ws1_f32[256 * 256 + ncol];   // distance coefficient row (fp32)
    float b1 = bs1[ncol], b2 = bt1[ncol], b3 = bi1[ncol];
    store_h_dist(Hs, s0, 0,  n0, lane, b1, dist_s, wl);
    store_h_dist(Hs, s1, 16, n0, lane, b1, dist_s, wl);
    store_h(Ht, t0, 0,  n0, lane, b2, true);
    store_h(Ht, t1, 16, n0, lane, b2, true);
    store_h(Hi, p0, 0,  n0, lane, b3, true);
    store_h(Hi, p1, 16, n0, lane, b3, true);
  }
  __syncthreads();

  // ---- layer 2 (no relu on second linear of _mlp2) ----
  {
    v8f s0 = {}, s1 = {}, t0 = {}, t1 = {}, p0 = {}, p1 = {};
#pragma unroll
    for (int kc = 0; kc < 8; ++kc) {
      int k0 = kc * 32;
      v16bf Bs  = loadB(W2s, k0, n0, lane);
      v16bf Bt  = loadB(W2t, k0, n0, lane);
      v16bf Bi  = loadB(W2i, k0, n0, lane);
      v16bf As0 = loadA(Hs, 0,  k0, lane);
      v16bf As1 = loadA(Hs, 16, k0, lane);
      v16bf At0 = loadA(Ht, 0,  k0, lane);
      v16bf At1 = loadA(Ht, 16, k0, lane);
      v16bf Ai0 = loadA(Hi, 0,  k0, lane);
      v16bf Ai1 = loadA(Hi, 16, k0, lane);
      s0 = wmma_bf16(As0, Bs, s0); s1 = wmma_bf16(As1, Bs, s1);
      t0 = wmma_bf16(At0, Bt, t0); t1 = wmma_bf16(At1, Bt, t1);
      p0 = wmma_bf16(Ai0, Bi, p0); p1 = wmma_bf16(Ai1, Bi, p1);
    }
    int ncol = n0 + (lane & 15);
    float b1 = bs2[ncol], b2 = bt2[ncol], b3 = bi2[ncol];
    store_h(Gs, s0, 0,  n0, lane, b1, false);
    store_h(Gs, s1, 16, n0, lane, b1, false);
    store_h(Gt, t0, 0,  n0, lane, b2, false);
    store_h(Gt, t1, 16, n0, lane, b2, false);
    store_h(Gi, p0, 0,  n0, lane, b3, false);
    store_h(Gi, p1, 16, n0, lane, b3, false);
  }
  __syncthreads();

  // ---- fusion: R = Gs@Wf[0:256] + Gt@Wf[256:512] + Gi@Wf[512:768] + bf ----
  {
    v8f r0 = {}, r1 = {};
#pragma unroll
    for (int kc = 0; kc < 8; ++kc) {
      int k0 = kc * 32;
      v16bf Ba  = loadB(WfA, k0, n0, lane);
      v16bf Bb  = loadB(WfB, k0, n0, lane);
      v16bf Bc  = loadB(WfC, k0, n0, lane);
      v16bf As0 = loadA(Gs, 0,  k0, lane);
      v16bf As1 = loadA(Gs, 16, k0, lane);
      v16bf At0 = loadA(Gt, 0,  k0, lane);
      v16bf At1 = loadA(Gt, 16, k0, lane);
      v16bf Ai0 = loadA(Gi, 0,  k0, lane);
      v16bf Ai1 = loadA(Gi, 16, k0, lane);
      r0 = wmma_bf16(As0, Ba, r0); r1 = wmma_bf16(As1, Ba, r1);
      r0 = wmma_bf16(At0, Bb, r0); r1 = wmma_bf16(At1, Bb, r1);
      r0 = wmma_bf16(Ai0, Bc, r0); r1 = wmma_bf16(Ai1, Bc, r1);
    }
    int n = lane & 15;
    int ncol = n0 + n;
    float bias = bfin[ncol];
    int moff = (lane >> 4) << 3;
    size_t rowbase = ((size_t)(b * 96 + i) * 96 + j0) * 256;
#pragma unroll
    for (int r = 0; r < 8; ++r) {
      int m = moff + r;
      // Streaming, write-once output: non-temporal stores keep L2 for weights.
      __builtin_nontemporal_store((r0[r] + bias) * mp_s[m],
                                  &out[rowbase + (size_t)m * 256 + ncol]);
      __builtin_nontemporal_store((r1[r] + bias) * mp_s[m + 16],
                                  &out[rowbase + (size_t)(m + 16) * 256 + ncol]);
    }
  }
}

extern "C" void kernel_launch(void* const* d_in, const int* in_sizes, int n_in,
                              void* d_out, int out_size, void* d_ws, size_t ws_size,
                              hipStream_t stream) {
  const float* features  = (const float*)d_in[0];
  const float* distances = (const float*)d_in[1];
  const float* mask      = (const float*)d_in[2];
  const float* Ws1 = (const float*)d_in[3];
  const float* bs1 = (const float*)d_in[4];
  const float* Ws2 = (const float*)d_in[5];
  const float* bs2 = (const float*)d_in[6];
  const float* Wt1 = (const float*)d_in[7];
  const float* bt1 = (const float*)d_in[8];
  const float* Wt2 = (const float*)d_in[9];
  const float* bt2 = (const float*)d_in[10];
  const float* Wi1 = (const float*)d_in[11];
  const float* bi1 = (const float*)d_in[12];
  const float* Wi2 = (const float*)d_in[13];
  const float* bi2 = (const float*)d_in[14];
  const float* Wf  = (const float*)d_in[15];
  const float* bfb = (const float*)d_in[16];

  __bf16* Wbf = (__bf16*)d_ws;   // 589,824 bf16 elems = 1.18 MB
  const int blk = 256;
  cvt_bf16<<<(65536 + blk - 1) / blk, blk, 0, stream>>>(Ws1, Wbf + 0,      65536);
  cvt_bf16<<<(65536 + blk - 1) / blk, blk, 0, stream>>>(Wt1, Wbf + 65536,  65536);
  cvt_bf16<<<(65536 + blk - 1) / blk, blk, 0, stream>>>(Wi1, Wbf + 131072, 65536);
  cvt_bf16<<<(65536 + blk - 1) / blk, blk, 0, stream>>>(Ws2, Wbf + 196608, 65536);
  cvt_bf16<<<(65536 + blk - 1) / blk, blk, 0, stream>>>(Wt2, Wbf + 262144, 65536);
  cvt_bf16<<<(65536 + blk - 1) / blk, blk, 0, stream>>>(Wi2, Wbf + 327680, 65536);
  cvt_bf16<<<(196608 + blk - 1) / blk, blk, 0, stream>>>(Wf, Wbf + 393216, 196608);

  size_t shmem = 768 + 6 * 8192 * sizeof(__bf16);  // 99,072 B < 320 KB/WGP
  relenc_wmma<<<8 * 96 * 3, 512, shmem, stream>>>(
      features, distances, mask, Ws1,
      bs1, bs2, bt1, bt2, bi1, bi2, bfb, Wbf, (float*)d_out);
}
